// Attention_13572096656114
// MI455X (gfx1250) — matrix operand order repaired and checked by reading, not compiled, not run on hardware
//
#include <hip/hip_runtime.h>

typedef __attribute__((ext_vector_type(16))) _Float16 v16h;
typedef __attribute__((ext_vector_type(8)))  _Float16 v8h;
typedef __attribute__((ext_vector_type(8)))  float    v8f;

#define B_   4
#define M_   1024
#define N_   1024
#define DX_  32
#define H_   32
#define DV_  64

#define KP   36   // k_lds row pitch in floats  (144B, 16B-aligned, conflict-free)
#define RP   40   // rT row pitch in halfs      (80B,  16B-aligned, conflict-free)

__global__ __launch_bounds__(128)
void lnp_laplace_attn_kernel(const float* __restrict__ x1,
                             const float* __restrict__ x2,
                             const float* __restrict__ r,
                             const float* __restrict__ Wm,
                             const float* __restrict__ bias,
                             float* __restrict__ out)
{
    __shared__ float    Wlds[H_ * DX_];
    __shared__ float    blds[H_];
    __shared__ float    klds[32 * KP];
    __shared__ _Float16 rT[DV_ * RP];

    const int tid  = threadIdx.x;
    const int lane = tid & 31;
    const int wave = tid >> 5;
    const int b    = blockIdx.x >> 4;          // 16 m-blocks per batch
    const int m0   = (blockIdx.x & 15) * 64;   // 64 m-rows per block

    // ---- stage W and bias in LDS (uniform-broadcast reads later) ----
    for (int i = tid; i < H_ * DX_; i += 128) Wlds[i] = Wm[i];
    if (tid < H_) blds[tid] = bias[tid];
    __syncthreads();

    // ---- per-lane query row: q[m][0..31], m = m0 + wave*16 + (lane&15) ----
    const int mrow = m0 + wave * 16 + (lane & 15);
    float xv[DX_];
    {
        const float4* xp = (const float4*)(x2 + ((size_t)b * M_ + mrow) * DX_);
        #pragma unroll
        for (int j = 0; j < 8; ++j) {
            float4 t = xp[j];
            xv[4*j] = t.x; xv[4*j+1] = t.y; xv[4*j+2] = t.z; xv[4*j+3] = t.w;
        }
    }
    float qv[H_];
    #pragma unroll 4
    for (int h = 0; h < H_; ++h) {
        float s = blds[h];
        #pragma unroll
        for (int d = 0; d < DX_; ++d) s += xv[d] * Wlds[h * DX_ + d];
        qv[h] = s;
    }

    v8f acc[4] = {};   // 16m x 64dv output tile per wave (4 WMMA accumulators)

    // cooperative k-chunk work split: thread -> (n-row, 8 h-values)
    const int kn  = tid >> 2;          // 0..31
    const int kh0 = (tid & 3) * 8;     // 0,8,16,24

    const int kbase_lane = (lane & 16) >> 1;   // A-layout: +8 K for upper half-wave
    const int nbase      = (lane & 16) >> 1;   // B-layout: K base 0 / 8
    const int dvcol      = lane & 15;          // B/C/D: N = lane & 15

    for (int n0 = 0; n0 < N_; n0 += 32) {
        // ---- prefetch next chunk's streaming rows (global_prefetch_b8) ----
        if (n0 + 32 < N_) {                       // uniform branch: EXEC stays full
            const size_t nrow = (size_t)b * N_ + n0 + 32 + kn;
            __builtin_prefetch(x1 + nrow * DX_, 0, 1);
            __builtin_prefetch(r  + nrow * DV_, 0, 1);
        }

        __syncthreads();   // previous iteration's LDS consumption complete

        // ---- compute k chunk [32n x 32h] into LDS ----
        {
            const float4* xp = (const float4*)(x1 + ((size_t)b * N_ + n0 + kn) * DX_);
            float xr[DX_];
            #pragma unroll
            for (int j = 0; j < 8; ++j) {
                float4 t = xp[j];
                xr[4*j] = t.x; xr[4*j+1] = t.y; xr[4*j+2] = t.z; xr[4*j+3] = t.w;
            }
            #pragma unroll
            for (int hh = 0; hh < 8; ++hh) {
                const int h = kh0 + hh;
                float s = blds[h];
                #pragma unroll
                for (int d = 0; d < DX_; ++d) s += xr[d] * Wlds[h * DX_ + d];
                klds[kn * KP + h] = s;
            }
        }

        // ---- stage r chunk transposed as f16: rT[dv][n] ----
        for (int idx = tid; idx < 32 * DV_; idx += 128) {
            const int n  = idx >> 6;       // 0..31
            const int dv = idx & 63;       // 0..63
            rT[dv * RP + n] = (_Float16)r[((size_t)b * N_ + n0 + n) * DV_ + dv];
        }
        __syncthreads();

        // ---- build A tile: weights in WMMA 16-bit A register layout ----
        // lane<16 : e in 0..7 -> K=e,   e in 8..15 -> K=e+8
        // lane>=16: same +8
        v16h a;
        #pragma unroll
        for (int e = 0; e < 16; ++e) {
            const int K = e + (e & 8) + kbase_lane;
            const float4* krow = (const float4*)&klds[K * KP];  // broadcast across 16 lanes
            float s = 0.0f;
            #pragma unroll
            for (int j = 0; j < 8; ++j) {
                float4 kv = krow[j];
                s += fabsf(kv.x - qv[4*j])   + fabsf(kv.y - qv[4*j+1])
                   + fabsf(kv.z - qv[4*j+2]) + fabsf(kv.w - qv[4*j+3]);
            }
            // 1 + tanh(-s) == 2 / (1 + exp(2s)):  v_exp_f32 + v_rcp_f32 (no IEEE div)
            const float wgt = 2.0f * __builtin_amdgcn_rcpf(1.0f + __expf(2.0f * s));
            a[e] = (_Float16)wgt;
        }

        // ---- 4 x v_wmma_f32_16x16x32_f16: acc[c] += A(16x32) * r(32x16) ----
        #pragma unroll
        for (int c = 0; c < 4; ++c) {
            const _Float16* bp = &rT[(c * 16 + dvcol) * RP + nbase];
            v8h lo = *(const v8h*)bp;          // K = nbase + 0..7
            v8h hi = *(const v8h*)(bp + 16);   // K = nbase + 16..23
            v16h bv;
            #pragma unroll
            for (int i = 0; i < 8; ++i) { bv[i] = lo[i]; bv[8 + i] = hi[i]; }
            acc[c] = __builtin_amdgcn_wmma_f32_16x16x32_f16(
                false, a, false, bv, (short)0, acc[c], false, false);
        }
    }

    // ---- store: C/D layout  M = v + 8*(lane>=16), N = lane&15 ----
    const int mbase = m0 + wave * 16 + ((lane & 16) >> 1);
    #pragma unroll
    for (int c = 0; c < 4; ++c) {
        #pragma unroll
        for (int v = 0; v < 8; ++v) {
            out[((size_t)b * M_ + mbase + v) * DV_ + c * 16 + dvcol] = acc[c][v];
        }
    }
}

extern "C" void kernel_launch(void* const* d_in, const int* in_sizes, int n_in,
                              void* d_out, int out_size, void* d_ws, size_t ws_size,
                              hipStream_t stream) {
    const float* x1   = (const float*)d_in[0];  // [B,N,DX]
    const float* x2   = (const float*)d_in[1];  // [B,M,DX]
    const float* r    = (const float*)d_in[2];  // [B,N,DV]
    const float* Wm   = (const float*)d_in[3];  // [H,DX]
    const float* bias = (const float*)d_in[4];  // [H]
    float* out = (float*)d_out;                 // [B,M,DV]

    dim3 grid(B_ * (M_ / 64));   // 64 blocks
    dim3 block(128);             // 4 waves, each owns a 16x64 output tile
    hipLaunchKernelGGL(lnp_laplace_attn_kernel, grid, block, 0, stream,
                       x1, x2, r, Wm, bias, out);
}
